// BaseGenerator_26053271617952
// MI455X (gfx1250) — compile-verified, matrix operand order checked by reading, not run on hardware
//
#include <hip/hip_runtime.h>

// ---------------------------------------------------------------------------
// MI455X (gfx1250, wave32) implementation of the BaseGenerator reference.
//  - All GEMMs: v_wmma_f32_16x16x32_bf16, f32 accumulate.
//  - Global->LDS staging: GLOBAL_LOAD_ASYNC_TO_LDS_B128 (ASYNCcnt-tracked),
//    double-buffered tiles so the TDM-path copy overlaps WMMA compute.
//  - All fp32 weights are pre-converted once to a bf16 pool so every GEMM
//    B operand is a uniform [N,K] row-major bf16 async-stageable tile.
//  - Attention bias (6 gathers + causal/pad mask) precomputed once as bf16
//    [B,H,S,S], reused by all 6 layers.
// ---------------------------------------------------------------------------

typedef __attribute__((ext_vector_type(16))) __bf16 v16bf;
typedef __attribute__((ext_vector_type(8)))  __bf16 v8bf;
typedef __attribute__((ext_vector_type(8)))  float  v8f;

#define DEV __device__ __forceinline__

static constexpr int Bn = 16, Sn = 256, En = 1024, Hn = 16, Fn = 4096;
static constexpr int ROWS = Bn * Sn;           // 4096
static constexpr int DH   = En / Hn;           // 64
static constexpr int GEN_OUT = 88, RSLOT = 40, EF_PAD = 48, VOCAB = 128;

static constexpr int BM = 128, BN = 128, BK = 32, LDT = 40; // LDS row pad: 80B

enum {
  EPI_QKV = 0,        // +bias, scatter to q[b,h,s,d], k[b,h,s,d], vT[b,h,d,s]
  EPI_BIAS_RES = 1,   // +bias[n] + res -> f32 (LN input)
  EPI_GELU = 2,       // gelu(+bias) -> bf16
  EPI_BIAS_BF16 = 3,  // +bias -> bf16
  EPI_SCORES = 4,     // *scale + attn_bias(bf16,-inf masked) -> f32
  EPI_PV = 5,         // -> o_bf[b,s, h*64+d]
  EPI_GEN = 6,        // +bias, pred-mask -> d_out cols [0,88)
  EPI_EDGE = 7        // *scale, pred-mask -> d_out cols [88,128)
};

struct GArgs {
  const __bf16* A;           // [M,K] row-major bf16 (per-z strided)
  const __bf16* B;           // [N,K] row-major bf16 (per-z strided)
  const float*  bias;        // [N] or nullptr
  const float*  res;         // residual [M,ldo]
  float*        out_f;
  __bf16*       out_bf;
  __bf16 *qo, *ko, *vto;     // EPI_QKV targets
  const __bf16* attn_bias;   // EPI_SCORES
  const unsigned char* mask; // pred_masks [ROWS,128]
  float scale;
  long a_zs, b_zs;           // z strides (elements)
  int M, N, K, lda, ldb, ldo;
};

// Async copy 16B from global to LDS; tracked by ASYNCcnt (no VGPR data path).
DEV void async_copy_b128(unsigned lds_off, const void* gaddr) {
  asm volatile("global_load_async_to_lds_b128 %0, %1, off"
               :: "v"(lds_off), "v"(gaddr)
               : "memory");
}

DEV void wait_asynccnt0() {
#if __has_builtin(__builtin_amdgcn_s_wait_asynccnt)
  __builtin_amdgcn_s_wait_asynccnt(0);
#else
  asm volatile("s_wait_asynccnt 0x0" ::: "memory");
#endif
}

// Load one 16x32 bf16 fragment from an LDS tile (row-major, stride LDT).
// ISA 7.12.2: lanes 0-15 hold rows 0-15 with K {0..7,16..23}; lanes 16-31
// hold the same rows with K {8..15,24..31}; pairs packed per VGPR.
DEV v16bf load_frag(const __bf16* tile, int row0, int lane) {
  int r  = row0 + (lane & 15);
  int kg = (lane >> 4) * 8;
  v8bf lo = *(const v8bf*)(tile + r * LDT + kg);
  v8bf hi = *(const v8bf*)(tile + r * LDT + kg + 16);
  return __builtin_shufflevector(lo, hi, 0,1,2,3,4,5,6,7,8,9,10,11,12,13,14,15);
}

DEV float gelu_exact(float x) {
  return 0.5f * x * (1.0f + erff(x * 0.70710678118654752f));
}

template <int EPI>
__global__ __launch_bounds__(256) void gemm_wmma(GArgs g) {
  __shared__ __bf16 As[2][BM * LDT];
  __shared__ __bf16 Bs[2][BN * LDT];

  const int tid = threadIdx.x;
  const int z   = blockIdx.z;
  const int nb  = blockIdx.x * BN;
  const int mb  = blockIdx.y * BM;

  const __bf16* Ap = g.A + (long)z * g.a_zs;
  const __bf16* Bp = g.B + (long)z * g.b_zs;

  v8f acc[4][2];
#pragma unroll
  for (int mi = 0; mi < 4; ++mi)
#pragma unroll
    for (int ni = 0; ni < 2; ++ni)
#pragma unroll
      for (int e = 0; e < 8; ++e) acc[mi][ni][e] = 0.0f;

  const int wid = tid >> 5, lane = tid & 31;
  const int wm = (wid >> 2) * 64;   // 2 waves tile M
  const int wn = (wid & 3) * 32;    // 4 waves tile N

  // Per-thread staging chunks: tile is 128x32 bf16 = 512 x 16B; 2 per thread.
  // Chunk c: row = c/4, col = (c%4)*8. B rows >= N are clamped in-bounds;
  // their (unused) products land in columns the epilogue never stores.
  auto issue_stage = [&](int k0, int buf) {
#pragma unroll
    for (int i = 0; i < 2; ++i) {
      int c = tid + i * 256;
      int r = c >> 2, col = (c & 3) * 8;
      const void* srcA = (const void*)(Ap + (long)(mb + r) * g.lda + k0 + col);
      unsigned dstA = (unsigned)(size_t)(&As[buf][r * LDT + col]);
      async_copy_b128(dstA, srcA);
      int gn = nb + r;
      int gnc = gn < g.N ? gn : g.N - 1;
      const void* srcB = (const void*)(Bp + (long)gnc * g.ldb + k0 + col);
      unsigned dstB = (unsigned)(size_t)(&Bs[buf][r * LDT + col]);
      async_copy_b128(dstB, srcB);
    }
  };

  const int nst = g.K / BK;
  issue_stage(0, 0);
  wait_asynccnt0();
  __syncthreads();

  for (int s = 0; s < nst; ++s) {
    const int cur = s & 1;
    if (s + 1 < nst) issue_stage((s + 1) * BK, cur ^ 1);

    v16bf afr[4], bfr[2];
#pragma unroll
    for (int mi = 0; mi < 4; ++mi) afr[mi] = load_frag(As[cur], wm + mi * 16, lane);
#pragma unroll
    for (int ni = 0; ni < 2; ++ni) bfr[ni] = load_frag(Bs[cur], wn + ni * 16, lane);
#pragma unroll
    for (int mi = 0; mi < 4; ++mi)
#pragma unroll
      for (int ni = 0; ni < 2; ++ni)
        acc[mi][ni] = __builtin_amdgcn_wmma_f32_16x16x32_bf16(
            false, afr[mi], false, bfr[ni], (short)0, acc[mi][ni], false, false);

    if (s + 1 < nst) {
      wait_asynccnt0();     // our async writes for stage s+1 are in LDS
      __syncthreads();      // everyone's are; also fences this stage's ds reads
    }
  }

  // ---- epilogue: C/D layout — vgpr r: lanes 0-15 row r, lanes 16-31 row r+8
  const int cn = lane & 15;
  const int r8 = (lane >> 4) * 8;
  const float NEG_INF = -__builtin_inff();
#pragma unroll
  for (int mi = 0; mi < 4; ++mi) {
#pragma unroll
    for (int ni = 0; ni < 2; ++ni) {
      int gn = nb + wn + ni * 16 + cn;
      if (gn >= g.N) continue;
      int gm0 = mb + wm + mi * 16 + r8;
#pragma unroll
      for (int r = 0; r < 8; ++r) {
        int gm = gm0 + r;
        float v = acc[mi][ni][r];
        if constexpr (EPI == EPI_QKV) {
          float t = v + g.bias[gn];
          __bf16 bv = (__bf16)t;
          int b = gm >> 8, s = gm & 255;
          if (gn < En) {
            int h = gn >> 6, d = gn & 63;
            g.qo[(long)((b * Hn + h) * Sn + s) * DH + d] = bv;
          } else if (gn < 2 * En) {
            int n2 = gn - En, h = n2 >> 6, d = n2 & 63;
            g.ko[(long)((b * Hn + h) * Sn + s) * DH + d] = bv;
          } else {
            int n2 = gn - 2 * En, h = n2 >> 6, d = n2 & 63;
            g.vto[(long)((b * Hn + h) * DH + d) * Sn + s] = bv;   // transposed
          }
        } else if constexpr (EPI == EPI_BIAS_RES) {
          long o = (long)gm * g.ldo + gn;
          g.out_f[o] = v + g.bias[gn] + g.res[o];
        } else if constexpr (EPI == EPI_GELU) {
          g.out_bf[(long)gm * g.ldo + gn] = (__bf16)gelu_exact(v + g.bias[gn]);
        } else if constexpr (EPI == EPI_BIAS_BF16) {
          g.out_bf[(long)gm * g.ldo + gn] = (__bf16)(v + g.bias[gn]);
        } else if constexpr (EPI == EPI_SCORES) {
          long o = (long)z * (Sn * Sn) + (long)gm * Sn + gn;
          g.out_f[o] = v * g.scale + (float)g.attn_bias[o];
        } else if constexpr (EPI == EPI_PV) {
          int b = z >> 4, h = z & 15;
          g.out_bf[(long)(b * Sn + gm) * En + h * DH + gn] = (__bf16)v;
        } else if constexpr (EPI == EPI_GEN) {
          long o = (long)gm * VOCAB + gn;
          g.out_f[o] = g.mask[o] ? NEG_INF : (v + g.bias[gn]);
        } else if constexpr (EPI == EPI_EDGE) {
          long o = (long)(z * Sn + gm) * VOCAB + GEN_OUT + gn;
          g.out_f[o] = g.mask[o] ? NEG_INF : v * g.scale;
        }
      }
    }
  }
}

// f32 -> bf16 bulk conversion (weights pool). n must be a multiple of 8.
__global__ __launch_bounds__(256) void f32_to_bf16_kernel(const float* src,
                                                          __bf16* dst, long n) {
  long t = ((long)blockIdx.x * 256 + threadIdx.x) * 8;
  if (t + 8 <= n) {
    float4 a = *(const float4*)(src + t);
    float4 b = *(const float4*)(src + t + 4);
    v8bf o;
    o[0] = (__bf16)a.x; o[1] = (__bf16)a.y; o[2] = (__bf16)a.z; o[3] = (__bf16)a.w;
    o[4] = (__bf16)b.x; o[5] = (__bf16)b.y; o[6] = (__bf16)b.z; o[7] = (__bf16)b.w;
    *(v8bf*)(dst + t) = o;
  }
}

// ---------------------------------------------------------------------------
// Attention bias: mean of 6 gathered [*,H] tables, causal + key-pad masked,
// stored bf16 as [B,H,S,S]. One block per (b,i) row; threads stride j.
// ---------------------------------------------------------------------------
__global__ __launch_bounds__(256) void build_bias_kernel(
    const int* dist, const int* up, const int* down, const int* bup,
    const int* bdown, const int* bright, const float* td, const float* tu,
    const float* tdn, const float* tbu, const float* tbd, const float* tbr,
    const int* seq, __bf16* bias) {
  int bi = blockIdx.x;                 // b*S + i
  int b = bi >> 8, i = bi & 255;
  int j = threadIdx.x;
  long base = (long)bi * Sn + j;
  int i0 = dist[base], i1 = up[base], i2 = down[base];
  int i3 = bup[base], i4 = bdown[base], i5 = bright[base];
  bool masked = (j > i) || (seq[b * Sn + j] == 0);
  const __bf16 NI = (__bf16)(-__builtin_inff());
#pragma unroll
  for (int h = 0; h < Hn; ++h) {
    float v = (td[i0 * Hn + h] + tu[i1 * Hn + h] + tdn[i2 * Hn + h] +
               tbu[i3 * Hn + h] + tbd[i4 * Hn + h] + tbr[i5 * Hn + h]) *
              (1.0f / 6.0f);
    long o = ((long)(b * Hn + h) * Sn + i) * Sn + j;
    bias[o] = masked ? NI : (__bf16)v;
  }
}

// x0 = (tok_emb[seq] + branch_emb[branch]) * sqrt(E)/2 ; write f32 + bf16.
__global__ __launch_bounds__(256) void embed_kernel(
    const int* seq, const int* br, const float* tok_emb,
    const float* branch_emb, float* xf, __bf16* xb) {
  long row = blockIdx.x;
  int t = seq[row], b = br[row];
#pragma unroll
  for (int i = 0; i < 4; ++i) {
    int c = threadIdx.x + i * 256;
    float v = (tok_emb[(long)t * En + c] + branch_emb[(long)b * En + c]) * 16.0f;
    xf[row * En + c] = v;
    xb[row * En + c] = (__bf16)v;
  }
}

// Row LayerNorm over E=1024; one block per row; writes f32 + bf16.
__global__ __launch_bounds__(256) void layernorm_kernel(
    const float* in, const float* gw, const float* bw, float* out_f,
    __bf16* out_bf) {
  __shared__ float red[256];
  long row = blockIdx.x;
  const float* x = in + row * En;
  int t = threadIdx.x;
  float vals[4], s = 0.0f, s2 = 0.0f;
#pragma unroll
  for (int i = 0; i < 4; ++i) {
    float v = x[t + i * 256];
    vals[i] = v; s += v; s2 += v * v;
  }
  red[t] = s; __syncthreads();
  for (int off = 128; off > 0; off >>= 1) {
    if (t < off) red[t] += red[t + off];
    __syncthreads();
  }
  float sum = red[0]; __syncthreads();
  red[t] = s2; __syncthreads();
  for (int off = 128; off > 0; off >>= 1) {
    if (t < off) red[t] += red[t + off];
    __syncthreads();
  }
  float sum2 = red[0];
  float mu = sum * (1.0f / En);
  float var = sum2 * (1.0f / En) - mu * mu;
  float inv = rsqrtf(var + 1e-5f);
#pragma unroll
  for (int i = 0; i < 4; ++i) {
    int c = t + i * 256;
    float y = (vals[i] - mu) * inv * gw[c] + bw[c];
    out_f[row * En + c] = y;
    out_bf[row * En + c] = (__bf16)y;
  }
}

// Row softmax over S=256 (-inf safe); scores f32 -> probs bf16.
__global__ __launch_bounds__(256) void softmax_kernel(const float* s, __bf16* p) {
  __shared__ float red[256];
  long row = blockIdx.x;
  int t = threadIdx.x;
  float v = s[row * Sn + t];
  red[t] = v; __syncthreads();
  for (int off = 128; off > 0; off >>= 1) {
    if (t < off) red[t] = fmaxf(red[t], red[t + off]);
    __syncthreads();
  }
  float mx = red[0]; __syncthreads();
  float e = __expf(v - mx);
  red[t] = e; __syncthreads();
  for (int off = 128; off > 0; off >>= 1) {
    if (t < off) red[t] += red[t + off];
    __syncthreads();
  }
  float inv = 1.0f / red[0];
  p[row * Sn + t] = (__bf16)(e * inv);
}

__global__ __launch_bounds__(256) void zero_bf16_kernel(__bf16* p, long n) {
  long t = (long)blockIdx.x * 256 + threadIdx.x;
  if (t < n) p[t] = (__bf16)0.0f;
}

// Scatter ring-token rows of out1_ into edge_feat[b, slot-1, :].
__global__ __launch_bounds__(256) void scatter_edge_kernel(
    const int* seq, const __bf16* out1, __bf16* ef) {
  long row = blockIdx.x;            // b*S + s
  int b = row >> 8;
  int sid = seq[row];
  int i = (sid < 88 || sid > 127) ? 87 : sid;
  int slot = i - 88 + 1;            // [0, 40]; 0 = dump slot (dropped)
  if (slot >= 1) {
    __bf16* d = ef + ((long)b * EF_PAD + (slot - 1)) * En;
    const __bf16* s = out1 + row * En;
#pragma unroll
    for (int q = 0; q < 4; ++q) {
      int c = threadIdx.x + q * 256;
      d[c] = s[c];
    }
  }
}

// ---------------------------------------------------------------------------
// Host orchestration
// ---------------------------------------------------------------------------
static inline dim3 gemm_grid(int M, int N, int Z) {
  return dim3((N + BN - 1) / BN, (M + BM - 1) / BM, Z);
}

extern "C" void kernel_launch(void* const* d_in, const int* in_sizes, int n_in,
                              void* d_out, int out_size, void* d_ws,
                              size_t ws_size, hipStream_t stream) {
  (void)in_sizes; (void)n_in; (void)out_size; (void)ws_size;

  // --- inputs (setup_inputs order), params flattened as a sorted pytree ---
  const int* seq    = (const int*)d_in[0];
  const int* branch = (const int*)d_in[1];
  const int* dist   = (const int*)d_in[2];
  const int* uploc  = (const int*)d_in[3];
  const int* dnloc  = (const int*)d_in[4];
  const int* buploc = (const int*)d_in[5];
  const int* bdnloc = (const int*)d_in[6];
  const int* brloc  = (const int*)d_in[7];
  const unsigned char* pred = (const unsigned char*)d_in[8];
  auto F = [&](int i) { return (const float*)d_in[i]; };
  const float* bdown_emb  = F(9);
  const float* branch_emb = F(10);
  const float* bright_emb = F(11);
  const float* bup_emb    = F(12);
  const float* dist_emb   = F(13);
  const float* down_emb   = F(14);
  const float* edge_W0 = F(15);
  const float* edge_W1 = F(16);
  const float* edge_b0 = F(17);
  const float* edge_b1 = F(18);
  const float* gen_W   = F(19);
  const float* gen_b   = F(20);
  const int LAYER0 = 21, LSTRIDE = 12;  // W1,W2,Wo,Wqkv,b1,b2,bo,bqkv,ln1_b,ln1_g,ln2_b,ln2_g
  const float* lnf_b   = F(LAYER0 + 6 * LSTRIDE + 0);
  const float* lnf_g   = F(LAYER0 + 6 * LSTRIDE + 1);
  const float* tok_emb = F(LAYER0 + 6 * LSTRIDE + 2);
  const float* up_emb  = F(LAYER0 + 6 * LSTRIDE + 3);

  float* out = (float*)d_out;

  // --- workspace bump allocator ---
  char* w = (char*)d_ws;
  auto alloc = [&](size_t bytes) {
    void* p = (void*)w;
    w += (bytes + 255) & ~(size_t)255;
    return p;
  };
  __bf16* bias_bf = (__bf16*)alloc((size_t)Bn * Hn * Sn * Sn * 2);
  float*  x_res   = (float*) alloc((size_t)ROWS * En * 4);
  __bf16* x_bf    = (__bf16*)alloc((size_t)ROWS * En * 2);
  __bf16* q_bf    = (__bf16*)alloc((size_t)Bn * Hn * Sn * DH * 2);
  __bf16* k_bf    = (__bf16*)alloc((size_t)Bn * Hn * Sn * DH * 2);
  __bf16* vT_bf   = (__bf16*)alloc((size_t)Bn * Hn * DH * Sn * 2);
  float*  scores  = (float*) alloc((size_t)Bn * Hn * Sn * Sn * 4);
  __bf16* p_bf    = (__bf16*)alloc((size_t)Bn * Hn * Sn * Sn * 2);
  __bf16* o_bf    = (__bf16*)alloc((size_t)ROWS * En * 2);
  float*  sub_f   = (float*) alloc((size_t)ROWS * En * 4);
  __bf16* h_bf    = (__bf16*)alloc((size_t)ROWS * Fn * 2);
  __bf16* out0_bf = (__bf16*)alloc((size_t)ROWS * En * 2);
  __bf16* out1_bf = (__bf16*)alloc((size_t)ROWS * En * 2);
  __bf16* ef_bf   = (__bf16*)alloc((size_t)Bn * EF_PAD * En * 2);

  // --- bf16 weight pool (converted once per call) ---
  const long QKV_N = 3L * En * En, SQ_N = (long)En * En, FF_N = (long)Fn * En;
  __bf16* Wqkv_bf[6]; __bf16* Wo_bf[6]; __bf16* W1_bf[6]; __bf16* W2_bf[6];
  for (int l = 0; l < 6; ++l) {
    Wqkv_bf[l] = (__bf16*)alloc((size_t)QKV_N * 2);
    Wo_bf[l]   = (__bf16*)alloc((size_t)SQ_N * 2);
    W1_bf[l]   = (__bf16*)alloc((size_t)FF_N * 2);
    W2_bf[l]   = (__bf16*)alloc((size_t)FF_N * 2);
  }
  __bf16* genW_bf = (__bf16*)alloc((size_t)GEN_OUT * En * 2);
  __bf16* eW0_bf  = (__bf16*)alloc((size_t)SQ_N * 2);
  __bf16* eW1_bf  = (__bf16*)alloc((size_t)SQ_N * 2);

  auto cvt = [&](const float* src, __bf16* dst, long n) {
    unsigned blocks = (unsigned)((n / 8 + 255) / 256);
    f32_to_bf16_kernel<<<blocks, 256, 0, stream>>>(src, dst, n);
  };
  for (int l = 0; l < 6; ++l) {
    cvt(F(LAYER0 + l * LSTRIDE + 3), Wqkv_bf[l], QKV_N);
    cvt(F(LAYER0 + l * LSTRIDE + 2), Wo_bf[l],   SQ_N);
    cvt(F(LAYER0 + l * LSTRIDE + 0), W1_bf[l],   FF_N);
    cvt(F(LAYER0 + l * LSTRIDE + 1), W2_bf[l],   FF_N);
  }
  cvt(gen_W,   genW_bf, (long)GEN_OUT * En);
  cvt(edge_W0, eW0_bf,  SQ_N);
  cvt(edge_W1, eW1_bf,  SQ_N);

  // --- precompute attention bias + embeddings ---
  build_bias_kernel<<<Bn * Sn, 256, 0, stream>>>(
      dist, uploc, dnloc, buploc, bdnloc, brloc, dist_emb, up_emb, down_emb,
      bup_emb, bdown_emb, bright_emb, seq, bias_bf);
  embed_kernel<<<ROWS, 256, 0, stream>>>(seq, branch, tok_emb, branch_emb,
                                         x_res, x_bf);

  for (int l = 0; l < 6; ++l) {
    const float* b1   = F(LAYER0 + l * LSTRIDE + 4);
    const float* b2   = F(LAYER0 + l * LSTRIDE + 5);
    const float* bo   = F(LAYER0 + l * LSTRIDE + 6);
    const float* bqkv = F(LAYER0 + l * LSTRIDE + 7);
    const float* ln1b = F(LAYER0 + l * LSTRIDE + 8);
    const float* ln1g = F(LAYER0 + l * LSTRIDE + 9);
    const float* ln2b = F(LAYER0 + l * LSTRIDE + 10);
    const float* ln2g = F(LAYER0 + l * LSTRIDE + 11);

    { // QKV projection: [4096,1024] x [3072,1024]^T
      GArgs g = {};
      g.A = x_bf; g.B = Wqkv_bf[l]; g.bias = bqkv;
      g.qo = q_bf; g.ko = k_bf; g.vto = vT_bf;
      g.M = ROWS; g.N = 3 * En; g.K = En; g.lda = En; g.ldb = En;
      gemm_wmma<EPI_QKV><<<gemm_grid(ROWS, 3 * En, 1), 256, 0, stream>>>(g);
    }
    { // scores = QK^T/8 + bias  (per b,h)
      GArgs g = {};
      g.A = q_bf; g.B = k_bf; g.attn_bias = bias_bf; g.out_f = scores;
      g.scale = 0.125f;  // 1/sqrt(64)
      g.a_zs = (long)Sn * DH; g.b_zs = (long)Sn * DH;
      g.M = Sn; g.N = Sn; g.K = DH; g.lda = DH; g.ldb = DH;
      gemm_wmma<EPI_SCORES><<<gemm_grid(Sn, Sn, Bn * Hn), 256, 0, stream>>>(g);
    }
    softmax_kernel<<<Bn * Hn * Sn, 256, 0, stream>>>(scores, p_bf);
    { // O = P @ V   (V pre-transposed to [N=dh, K=S])
      GArgs g = {};
      g.A = p_bf; g.B = vT_bf; g.out_bf = o_bf;
      g.a_zs = (long)Sn * Sn; g.b_zs = (long)DH * Sn;
      g.M = Sn; g.N = DH; g.K = Sn; g.lda = Sn; g.ldb = Sn;
      gemm_wmma<EPI_PV><<<gemm_grid(Sn, DH, Bn * Hn), 256, 0, stream>>>(g);
    }
    { // attn out proj + residual
      GArgs g = {};
      g.A = o_bf; g.B = Wo_bf[l]; g.bias = bo; g.res = x_res; g.out_f = sub_f;
      g.M = ROWS; g.N = En; g.K = En; g.lda = En; g.ldb = En; g.ldo = En;
      gemm_wmma<EPI_BIAS_RES><<<gemm_grid(ROWS, En, 1), 256, 0, stream>>>(g);
    }
    layernorm_kernel<<<ROWS, 256, 0, stream>>>(sub_f, ln1g, ln1b, x_res, x_bf);
    { // FFN up + exact GELU
      GArgs g = {};
      g.A = x_bf; g.B = W1_bf[l]; g.bias = b1; g.out_bf = h_bf;
      g.M = ROWS; g.N = Fn; g.K = En; g.lda = En; g.ldb = En; g.ldo = Fn;
      gemm_wmma<EPI_GELU><<<gemm_grid(ROWS, Fn, 1), 256, 0, stream>>>(g);
    }
    { // FFN down + residual
      GArgs g = {};
      g.A = h_bf; g.B = W2_bf[l]; g.bias = b2; g.res = x_res; g.out_f = sub_f;
      g.M = ROWS; g.N = En; g.K = Fn; g.lda = Fn; g.ldb = Fn; g.ldo = En;
      gemm_wmma<EPI_BIAS_RES><<<gemm_grid(ROWS, En, 1), 256, 0, stream>>>(g);
    }
    layernorm_kernel<<<ROWS, 256, 0, stream>>>(sub_f, ln2g, ln2b, x_res, x_bf);
  }

  // final LN (in/out alias is safe: each block reads its row before writing)
  layernorm_kernel<<<ROWS, 256, 0, stream>>>(x_res, lnf_g, lnf_b, x_res, x_bf);

  { // generator head -> d_out[:, 0:88], pred-masked
    GArgs g = {};
    g.A = x_bf; g.B = genW_bf; g.bias = gen_b; g.out_f = out; g.mask = pred;
    g.M = ROWS; g.N = GEN_OUT; g.K = En; g.lda = En; g.ldb = En;
    gemm_wmma<EPI_GEN><<<gemm_grid(ROWS, GEN_OUT, 1), 256, 0, stream>>>(g);
  }
  { // out0 = x @ W0^T + b0
    GArgs g = {};
    g.A = x_bf; g.B = eW0_bf; g.bias = edge_b0; g.out_bf = out0_bf;
    g.M = ROWS; g.N = En; g.K = En; g.lda = En; g.ldb = En; g.ldo = En;
    gemm_wmma<EPI_BIAS_BF16><<<gemm_grid(ROWS, En, 1), 256, 0, stream>>>(g);
  }
  { // out1_ = x @ W1^T + b1
    GArgs g = {};
    g.A = x_bf; g.B = eW1_bf; g.bias = edge_b1; g.out_bf = out1_bf;
    g.M = ROWS; g.N = En; g.K = En; g.lda = En; g.ldb = En; g.ldo = En;
    gemm_wmma<EPI_BIAS_BF16><<<gemm_grid(ROWS, En, 1), 256, 0, stream>>>(g);
  }
  {
    long n = (long)Bn * EF_PAD * En;
    zero_bf16_kernel<<<(unsigned)((n + 255) / 256), 256, 0, stream>>>(ef_bf, n);
  }
  scatter_edge_kernel<<<ROWS, 256, 0, stream>>>(seq, out1_bf, ef_bf);
  { // edge logits = E^-1/2 * out0 @ edge_feat^T -> d_out[:, 88:128]
    GArgs g = {};
    g.A = out0_bf; g.B = ef_bf; g.out_f = out; g.mask = pred;
    g.scale = 0.03125f;  // 1/sqrt(1024)
    g.a_zs = (long)Sn * En; g.b_zs = (long)EF_PAD * En;
    g.M = Sn; g.N = RSLOT; g.K = En; g.lda = En; g.ldb = En;
    gemm_wmma<EPI_EDGE><<<gemm_grid(Sn, RSLOT, Bn), 256, 0, stream>>>(g);
  }
}
